// motion_ae_17463337026115
// MI455X (gfx1250) — matrix operand order repaired and checked by reading, not compiled, not run on hardware
//
#include <hip/hip_runtime.h>
#include <hip/hip_bf16.h>

// Problem constants (from the reference): B=512, T=128, H=1024.
#define B_  512
#define T_  128
#define H_  1024
#define G3_ 3072   // 3*H

typedef __hip_bfloat16 bf16;
typedef __attribute__((ext_vector_type(8)))  __bf16 v8bf;
typedef __attribute__((ext_vector_type(16))) __bf16 v16bf;
typedef __attribute__((ext_vector_type(8)))  float  v8f;

__device__ __forceinline__ v16bf cat8(v8bf lo, v8bf hi) {
  return __builtin_shufflevector(lo, hi, 0,1,2,3,4,5,6,7,8,9,10,11,12,13,14,15);
}

// One 16-byte global->LDS async copy (CDNA5 TDM-lite path, tracked by ASYNCcnt).
// lds_off: raw 32-bit LDS byte address (low 32 bits of a generic shared pointer).
__device__ __forceinline__ void async_cp16(unsigned lds_off, const void* gaddr) {
  asm volatile("global_load_async_to_lds_b128 %0, %1, off"
               :: "v"(lds_off), "v"(gaddr) : "memory");
}

// ---------------------------------------------------------------------------
// bf16 WMMA GEMM:  C[M x N] (f32) = A[M x K] (bf16, row-major, lda)
//                                 @ W[N x K]^T (bf16, row-major, ldb)
// Block: 256 threads = 8 waves as 2(M) x 4(N); wave tile 32x64; WG tile 64x256.
// A/B tiles (4KB / 16KB per 32-K step) are staged into LDS with
// global_load_async_to_lds_b128, double-buffered; fragments read via ds_load.
// grid = (M/64, N/256). K multiple of 32.
// ---------------------------------------------------------------------------
__global__ __launch_bounds__(256) void gemm_bf16_wmma(
    const bf16* __restrict__ Ab, int lda,
    const bf16* __restrict__ Wb, int ldb,
    float* __restrict__ C, int ldc, int K)
{
  __shared__ __align__(16) __bf16 sA[2][64 * 32];    // 2 x 4KB
  __shared__ __align__(16) __bf16 sB[2][256 * 32];   // 2 x 16KB

  const int tid  = threadIdx.x;
  const int lane = tid & 31;
  const int wave = tid >> 5;
  const int wm   = wave >> 2;        // 0..1
  const int wn   = wave & 3;         // 0..3
  const int row  = lane & 15;
  const int sel  = lane >> 4;        // 0 or 1 (lane half)
  const int m0b  = blockIdx.x * 64;
  const int n0b  = blockIdx.y * 256;
  const int m0   = m0b + wm * 32;
  const int n0   = n0b + wn * 64;

  const __bf16* A = reinterpret_cast<const __bf16*>(Ab);
  const __bf16* W = reinterpret_cast<const __bf16*>(Wb);

  // Per-thread staging addresses (constant across K-tiles except the k term).
  const int ar = tid >> 2, aq = tid & 3;                 // A: chunk (row, 16B-quad)
  const __bf16* gA0 = A + (size_t)(m0b + ar) * lda + aq * 8;
  const __bf16* gB0 = W + (size_t)(n0b + tid) * ldb;     // B: one 64B row per thread
  const unsigned lA[2] = {
    (unsigned)(unsigned long long)(&sA[0][ar * 32 + aq * 8]),
    (unsigned)(unsigned long long)(&sA[1][ar * 32 + aq * 8]) };
  const unsigned lB[2] = {
    (unsigned)(unsigned long long)(&sB[0][tid * 32]),
    (unsigned)(unsigned long long)(&sB[1][tid * 32]) };

  const v8f vz = {0.f,0.f,0.f,0.f,0.f,0.f,0.f,0.f};
  v8f acc[2][4];
#pragma unroll
  for (int i = 0; i < 2; ++i)
#pragma unroll
    for (int j = 0; j < 4; ++j) acc[i][j] = vz;

  // ISA 16-bit A 16x32 layout: lanes 0-15 hold K[0..7]+K[16..23],
  // lanes 16-31 hold K[8..15]+K[24..31]  -> two 16B chunks at koffA, koffA+16.
  // B 32x16 layout (matching A): lanes 0-15 hold K[0..7]+K[16..23],
  // lanes 16-31 hold K[8..15]+K[24..31] -> two 16B chunks at koffB, koffB+16.
  const int koffA = sel * 8;
  const int koffB = sel * 8;

  const int nk = K >> 5;

  // Prologue: stage tile 0 into buffer 0 (5 async b128 per thread).
  async_cp16(lA[0], gA0);
#pragma unroll
  for (int q = 0; q < 4; ++q) async_cp16(lB[0] + q * 16, gB0 + q * 8);

  for (int kt = 0; kt < nk; ++kt) {
    const int cur = kt & 1;
    if (kt + 1 < nk) {
      const int k1 = (kt + 1) << 5;
      async_cp16(lA[cur ^ 1], gA0 + k1);
#pragma unroll
      for (int q = 0; q < 4; ++q) async_cp16(lB[cur ^ 1] + q * 16, gB0 + k1 + q * 8);
      // In-order completion: <=5 outstanding => tile kt fully in LDS.
      asm volatile("s_wait_asynccnt 0x5" ::: "memory");
    } else {
      asm volatile("s_wait_asynccnt 0x0" ::: "memory");
    }
    __syncthreads();   // tile kt visible to all waves

    v16bf af[2], bfr[4];
#pragma unroll
    for (int mi = 0; mi < 2; ++mi) {
      const __bf16* p = &sA[cur][(wm * 32 + mi * 16 + row) * 32 + koffA];
      af[mi] = cat8(*(const v8bf*)p, *(const v8bf*)(p + 16));
    }
#pragma unroll
    for (int ni = 0; ni < 4; ++ni) {
      const __bf16* q = &sB[cur][(wn * 64 + ni * 16 + row) * 32 + koffB];
      bfr[ni] = cat8(*(const v8bf*)q, *(const v8bf*)(q + 16));
    }
#pragma unroll
    for (int mi = 0; mi < 2; ++mi)
#pragma unroll
      for (int ni = 0; ni < 4; ++ni)
        acc[mi][ni] = __builtin_amdgcn_wmma_f32_16x16x32_bf16(
            false, af[mi], false, bfr[ni], (short)0, acc[mi][ni], false, false);

    __syncthreads();   // everyone done reading before buffer is overwritten
  }

  // C 16x16 f32 layout: VGPR i -> M = i + sel*8, N = lane&15.
#pragma unroll
  for (int mi = 0; mi < 2; ++mi)
#pragma unroll
    for (int ni = 0; ni < 4; ++ni) {
      const int n = n0 + ni * 16 + row;
      float* cp = C + (size_t)(m0 + mi * 16 + sel * 8) * ldc + n;
#pragma unroll
      for (int i = 0; i < 8; ++i) cp[(size_t)i * ldc] = acc[mi][ni][i];
    }
}

// ---------------------------------------------------------------------------
// Elementwise / small kernels
// ---------------------------------------------------------------------------
__global__ void f32_to_bf16_k(const float* __restrict__ s, bf16* __restrict__ d, int n) {
  int i = blockIdx.x * blockDim.x + threadIdx.x;
  if (i < n) d[i] = __float2bfloat16(s[i]);
}

__global__ void zero_bf16_k(bf16* d, int n) {
  int i = blockIdx.x * blockDim.x + threadIdx.x;
  if (i < n) d[i] = __float2bfloat16(0.f);
}

__global__ void zero_f32_k(float* d, int n) {
  int i = blockIdx.x * blockDim.x + threadIdx.x;
  if (i < n) d[i] = 0.f;
}

// obs_enc_t = relu(obs[:,t,:] @ enc_fc_w.T + enc_fc_b) -> bf16 into A cols [0,512)
__global__ void obs_enc_k(const float* __restrict__ obs, int t,
                          const float* __restrict__ w, const float* __restrict__ b,
                          bf16* __restrict__ A) {
  int i = blockIdx.x * blockDim.x + threadIdx.x;   // B_*512 threads
  int bb = i >> 9;
  int j  = i & 511;
  const float* o = obs + ((size_t)bb * T_ + t) * 4;
  const float* wr = w + (size_t)j * 4;
  float v = b[j] + o[0]*wr[0] + o[1]*wr[1] + o[2]*wr[2] + o[3]*wr[3];
  v = fmaxf(v, 0.f);
  A[(size_t)bb * 1536 + j] = __float2bfloat16(v);
}

__device__ __forceinline__ float sigmoidf_(float x) { return 1.f / (1.f + __expf(-x)); }

// Encoder GRU gate math; writes h (f32) and bf16 h into A cols [512,1536)
__global__ void gru_enc_k(const float* __restrict__ gi, const float* __restrict__ gh,
                          const float* __restrict__ bih, const float* __restrict__ bhh,
                          float* __restrict__ h, bf16* __restrict__ A) {
  int i = blockIdx.x * blockDim.x + threadIdx.x;   // B_*H_ threads
  int bb = i >> 10;
  int j  = i & 1023;
  size_t base = (size_t)bb * G3_;
  float gir = gi[base + j]        + bih[j];
  float giz = gi[base + H_  + j]  + bih[H_  + j];
  float gin = gi[base + 2*H_ + j] + bih[2*H_ + j];
  float ghr = gh[base + j]        + bhh[j];
  float ghz = gh[base + H_  + j]  + bhh[H_  + j];
  float ghn = gh[base + 2*H_ + j] + bhh[2*H_ + j];
  float r = sigmoidf_(gir + ghr);
  float z = sigmoidf_(giz + ghz);
  float n = tanhf(gin + r * ghn);
  float hv = h[(size_t)bb * H_ + j];
  float hn = (1.f - z) * n + z * hv;
  h[(size_t)bb * H_ + j] = hn;
  A[(size_t)bb * 1536 + 512 + j] = __float2bfloat16(hn);
}

// Decoder GRU: gi computed inline from 4-wide input (mask-selected), gh from GEMM.
__global__ void gru_dec_k(const float* __restrict__ gh, const float* __restrict__ obs,
                          int t, int first, const unsigned char* __restrict__ mask,
                          const float* __restrict__ x,
                          const float* __restrict__ Wih, const float* __restrict__ bih,
                          const float* __restrict__ bhh,
                          float* __restrict__ h, bf16* __restrict__ A) {
  int i = blockIdx.x * blockDim.x + threadIdx.x;   // B_*H_ threads
  int bb = i >> 10;
  int j  = i & 1023;
  bool useObs = (!first) && (mask[t - 1] != 0);   // mask[t-1] when t>=1
  const float* src = useObs ? (obs + ((size_t)bb * T_ + t) * 4) : (x + (size_t)bb * 4);
  float in0 = src[0], in1 = src[1], in2 = src[2], in3 = src[3];
  const float* wr = Wih + (size_t)(j)        * 4;
  const float* wz = Wih + (size_t)(H_  + j)  * 4;
  const float* wn = Wih + (size_t)(2*H_ + j) * 4;
  float gir = bih[j]        + in0*wr[0] + in1*wr[1] + in2*wr[2] + in3*wr[3];
  float giz = bih[H_  + j]  + in0*wz[0] + in1*wz[1] + in2*wz[2] + in3*wz[3];
  float gin = bih[2*H_ + j] + in0*wn[0] + in1*wn[1] + in2*wn[2] + in3*wn[3];
  size_t base = (size_t)bb * G3_;
  float ghr = gh[base + j]        + bhh[j];
  float ghz = gh[base + H_  + j]  + bhh[H_  + j];
  float ghn = gh[base + 2*H_ + j] + bhh[2*H_ + j];
  float r = sigmoidf_(gir + ghr);
  float z = sigmoidf_(giz + ghz);
  float n = tanhf(gin + r * ghn);
  float hv = h[(size_t)bb * H_ + j];
  float hn = (1.f - z) * n + z * hv;
  h[(size_t)bb * H_ + j] = hn;
  A[(size_t)bb * 1536 + 512 + j] = __float2bfloat16(hn);
}

__global__ void init_x_k(const float* __restrict__ obs, float* __restrict__ x) {
  int i = blockIdx.x * blockDim.x + threadIdx.x;   // B_*4 threads
  int bb = i >> 2;
  int o  = i & 3;
  x[(size_t)bb * 4 + o] = obs[((size_t)bb * T_ + 0) * 4 + o];  // init_motion
}

// x_new = relu(y + fc1_b) @ fc2_w.T + fc2_b + x_prev ; store to out[:, t, :] and x.
__global__ void fc2_k(const float* __restrict__ y, const float* __restrict__ b1,
                      const float* __restrict__ w2, const float* __restrict__ b2,
                      float* __restrict__ x, float* __restrict__ out, int t) {
  int i = blockIdx.x * blockDim.x + threadIdx.x;   // B_*4 threads
  int bb = i >> 2;
  int o  = i & 3;
  const float* yr = y + (size_t)bb * 256;
  const float* wr = w2 + (size_t)o * 256;
  float acc = b2[o];
  for (int j = 0; j < 256; ++j) acc += fmaxf(yr[j] + b1[j], 0.f) * wr[j];
  float xn = acc + x[(size_t)bb * 4 + o];
  x[(size_t)bb * 4 + o] = xn;
  out[((size_t)bb * T_ + t) * 4 + o] = xn;
}

// ---------------------------------------------------------------------------
// Launch
// ---------------------------------------------------------------------------
extern "C" void kernel_launch(void* const* d_in, const int* in_sizes, int n_in,
                              void* d_out, int out_size, void* d_ws, size_t ws_size,
                              hipStream_t stream) {
  (void)in_sizes; (void)n_in; (void)out_size; (void)ws_size;
  const float*         obs      = (const float*)d_in[0];
  const unsigned char* mask     = (const unsigned char*)d_in[1];  // jax bool -> 1 byte
  const float*         enc_fc_w = (const float*)d_in[2];
  const float*         enc_fc_b = (const float*)d_in[3];
  const float*         enc_Wih  = (const float*)d_in[4];
  const float*         enc_Whh  = (const float*)d_in[5];
  const float*         enc_bih  = (const float*)d_in[6];
  const float*         enc_bhh  = (const float*)d_in[7];
  const float*         dec_Wih  = (const float*)d_in[8];
  const float*         dec_Whh  = (const float*)d_in[9];
  const float*         dec_bih  = (const float*)d_in[10];
  const float*         dec_bhh  = (const float*)d_in[11];
  const float*         fc1_w    = (const float*)d_in[12];
  const float*         fc1_b    = (const float*)d_in[13];
  const float*         fc2_w    = (const float*)d_in[14];
  const float*         fc2_b    = (const float*)d_in[15];
  float* out = (float*)d_out;

  // Workspace carving (256B aligned).
  char* p = (char*)d_ws;
  auto alloc = [&](size_t bytes) { char* r = p; p += (bytes + 255) & ~(size_t)255; return r; };
  bf16*  Wih_e = (bf16*)alloc((size_t)G3_ * 512  * 2);
  bf16*  Whh_e = (bf16*)alloc((size_t)G3_ * H_   * 2);
  bf16*  Whh_d = (bf16*)alloc((size_t)G3_ * H_   * 2);
  bf16*  W1    = (bf16*)alloc((size_t)256 * H_   * 2);
  bf16*  Abuf  = (bf16*)alloc((size_t)B_  * 1536 * 2);   // [obs_enc | h] bf16, lda=1536
  float* gi    = (float*)alloc((size_t)B_ * G3_ * 4);
  float* gh    = (float*)alloc((size_t)B_ * G3_ * 4);
  float* h     = (float*)alloc((size_t)B_ * H_  * 4);
  float* y     = (float*)alloc((size_t)B_ * 256 * 4);
  float* x     = (float*)alloc((size_t)B_ * 4   * 4);

  const dim3 blk(256);

  // One-time weight conversion to bf16 + state init.
  f32_to_bf16_k<<<(G3_*512 + 255)/256, blk, 0, stream>>>(enc_Wih, Wih_e, G3_*512);
  f32_to_bf16_k<<<(G3_*H_  + 255)/256, blk, 0, stream>>>(enc_Whh, Whh_e, G3_*H_);
  f32_to_bf16_k<<<(G3_*H_  + 255)/256, blk, 0, stream>>>(dec_Whh, Whh_d, G3_*H_);
  f32_to_bf16_k<<<(256*H_  + 255)/256, blk, 0, stream>>>(fc1_w,   W1,    256*H_);
  zero_bf16_k<<<(B_*1536 + 255)/256, blk, 0, stream>>>(Abuf, B_*1536);
  zero_f32_k<<<(B_*H_ + 255)/256, blk, 0, stream>>>(h, B_*H_);

  const dim3 gGates(B_/64, G3_/256);  // (8, 12)
  const dim3 gFc1  (B_/64, 256/256);  // (8, 1)

  // Encoder scan.
  for (int t = 0; t < T_; ++t) {
    obs_enc_k<<<(B_*512)/256, blk, 0, stream>>>(obs, t, enc_fc_w, enc_fc_b, Abuf);
    gemm_bf16_wmma<<<gGates, blk, 0, stream>>>(Abuf,       1536, Wih_e, 512,  gi, G3_, 512);
    gemm_bf16_wmma<<<gGates, blk, 0, stream>>>(Abuf + 512, 1536, Whh_e, H_,   gh, G3_, H_);
    gru_enc_k<<<(B_*H_)/256, blk, 0, stream>>>(gi, gh, enc_bih, enc_bhh, h, Abuf);
  }

  // Decoder scan (h and its bf16 copy carry over from the encoder).
  init_x_k<<<(B_*4)/256, blk, 0, stream>>>(obs, x);
  for (int t = 0; t < T_; ++t) {
    gemm_bf16_wmma<<<gGates, blk, 0, stream>>>(Abuf + 512, 1536, Whh_d, H_, gh, G3_, H_);
    gru_dec_k<<<(B_*H_)/256, blk, 0, stream>>>(gh, obs, t, (t == 0) ? 1 : 0, mask, x,
                                               dec_Wih, dec_bih, dec_bhh, h, Abuf);
    gemm_bf16_wmma<<<gFc1, blk, 0, stream>>>(Abuf + 512, 1536, W1, H_, y, 256, H_);
    fc2_k<<<(B_*4)/256, blk, 0, stream>>>(y, fc1_b, fc2_w, fc2_b, x, out, t);
  }
}
